// FRCNN_45380624449835
// MI455X (gfx1250) — matrix operand order repaired and checked by reading, not compiled, not run on hardware
//
#include <hip/hip_runtime.h>
#include <hip/hip_bf16.h>
#include <math.h>

typedef __attribute__((ext_vector_type(16))) __bf16 v16bf;
typedef __attribute__((ext_vector_type(8)))  float  v8f;

#define FH 37
#define FW 37
#define NPIX (FH*FW)          // 1369
#define CIN 512
#define NANCH (NPIX*9)        // 12321
#define NSORT 16384
#define PRE_NMS_N 6000
#define POST_NMS_N 300
#define KFC1 25088
#define NFC 4096
#define MPAD 304              // 300 rois padded to 19*16

// ---------------------------------------------------------------- conv 3x3 + relu
__global__ void conv3x3_relu_kernel(const float* __restrict__ F,
                                    const float* __restrict__ W,
                                    const float* __restrict__ B,
                                    float* __restrict__ X) {
  __shared__ float patch[CIN * 9];
  int p = blockIdx.x;
  int py = p / FW, px = p % FW;
  for (int idx = threadIdx.x; idx < CIN * 9; idx += 256) {
    int ic = idx / 9, tap = idx % 9;
    int ky = tap / 3, kx = tap % 3;
    int iy = py + ky - 1, ix = px + kx - 1;
    float v = 0.f;
    if (iy >= 0 && iy < FH && ix >= 0 && ix < FW) v = F[(size_t)ic * NPIX + iy * FW + ix];
    patch[idx] = v;
  }
  __syncthreads();
  for (int rep = 0; rep < 2; ++rep) {
    int oc = threadIdx.x + rep * 256;
    float acc = B[oc];
    const float* wbase = W + (size_t)oc * CIN * 9;
    for (int ic = 0; ic < CIN; ++ic) {
      const float* pp = &patch[ic * 9];
      const float* wp = &wbase[ic * 9];
#pragma unroll
      for (int t = 0; t < 9; ++t) acc = fmaf(pp[t], wp[t], acc);
    }
    X[(size_t)oc * NPIX + p] = fmaxf(acc, 0.f);
  }
}

// ---------------------------------------------------------- RPN 1x1 convs + anchor decode
__global__ void rpn_kernel(const float* __restrict__ X,
                           const float* __restrict__ Wc, const float* __restrict__ Bc,
                           const float* __restrict__ Wr, const float* __restrict__ Br,
                           float* __restrict__ boxes, float* __restrict__ scores) {
  __shared__ float xc[CIN];
  __shared__ float outv[54];
  int p = blockIdx.x;
  for (int i = threadIdx.x; i < CIN; i += 64) xc[i] = X[(size_t)i * NPIX + p];
  __syncthreads();
  int t = threadIdx.x;
  if (t < 54) {
    const float* w; float b;
    if (t < 36) { w = Wr + (size_t)t * CIN; b = Br[t]; }
    else        { w = Wc + (size_t)(t - 36) * CIN; b = Bc[t - 36]; }
    float acc = b;
    for (int i = 0; i < CIN; ++i) acc = fmaf(xc[i], w[i], acc);
    outv[t] = acc;
  }
  __syncthreads();
  if (t < 9) {
    const float ratios[3] = {0.5f, 1.f, 2.f};
    const float scales[3] = {8.f, 16.f, 32.f};
    int r = t / 3, s = t % 3;
    float ah = 16.f * scales[s] * sqrtf(ratios[r]);
    float aw = 16.f * scales[s] * sqrtf(1.f / ratios[r]);
    int py = p / FW, px = p % FW;
    const float inv = 1.f / 600.f;
    float acx = (px * 16.f + 8.f) * inv, acy = (py * 16.f + 8.f) * inv;
    float awn = aw * inv, ahn = ah * inv;
    float r0 = outv[4 * t + 0], r1 = outv[4 * t + 1], r2 = outv[4 * t + 2], r3 = outv[4 * t + 3];
    float c0 = outv[36 + 2 * t], c1 = outv[36 + 2 * t + 1];
    float fg = 1.f / (1.f + expf(c0 - c1));            // softmax fg prob
    float ncx = r0 * awn + acx, ncy = r1 * ahn + acy;
    float nw = expf(r2) * awn, nh = expf(r3) * ahn;
    float x1 = fminf(fmaxf(ncx - nw * 0.5f, 0.f), 1.f);
    float y1 = fminf(fmaxf(ncy - nh * 0.5f, 0.f), 1.f);
    float x2 = fminf(fmaxf(ncx + nw * 0.5f, 0.f), 1.f);
    float y2 = fminf(fmaxf(ncy + nh * 0.5f, 0.f), 1.f);
    bool keep = ((x2 - x1) >= 0.016f) && ((y2 - y1) >= 0.016f);
    int a = p * 9 + t;
    boxes[a * 4 + 0] = keep ? x1 : 0.f;
    boxes[a * 4 + 1] = keep ? y1 : 0.f;
    boxes[a * 4 + 2] = keep ? x2 : 0.f;
    boxes[a * 4 + 3] = keep ? y2 : 0.f;
    scores[a] = keep ? fg : -1e9f;
  }
}

// -------------------------------------------------- bitonic sort (desc) in workspace memory
__global__ __launch_bounds__(1024) void sort_kernel(const float* __restrict__ scores,
                                                    const float* __restrict__ boxes,
                                                    float* __restrict__ key,
                                                    int* __restrict__ val,
                                                    float* __restrict__ sbox) {
  int tid = threadIdx.x;
  for (int i = tid; i < NSORT; i += 1024) {
    key[i] = (i < NANCH) ? scores[i] : -INFINITY;
    val[i] = (i < NANCH) ? i : 0;
  }
  __syncthreads();
  for (unsigned k = 2; k <= NSORT; k <<= 1) {
    for (unsigned j = k >> 1; j > 0; j >>= 1) {
      for (unsigned i = tid; i < NSORT; i += 1024) {
        unsigned ixj = i ^ j;
        if (ixj > i) {
          float a = key[i], b = key[ixj];
          bool desc = (i & k) == 0;
          if (desc ? (a < b) : (a > b)) {
            key[i] = b; key[ixj] = a;
            int va = val[i]; val[i] = val[ixj]; val[ixj] = va;
          }
        }
      }
      __syncthreads();
    }
  }
  for (int i = tid; i < PRE_NMS_N; i += 1024) {
    int idx = val[i];
    sbox[i * 4 + 0] = boxes[idx * 4 + 0];
    sbox[i * 4 + 1] = boxes[idx * 4 + 1];
    sbox[i * 4 + 2] = boxes[idx * 4 + 2];
    sbox[i * 4 + 3] = boxes[idx * 4 + 3];
  }
}

// ----------------------------------------------------------------- sequential greedy NMS
__global__ __launch_bounds__(1024) void nms_kernel(const float* __restrict__ b,
                                                   int* __restrict__ keep,
                                                   float* __restrict__ rois) {
  int tid = threadIdx.x;
  for (int i = tid; i < PRE_NMS_N; i += 1024) keep[i] = 1;
  __syncthreads();
  for (int i = 0; i < PRE_NMS_N; ++i) {
    if (keep[i]) {                       // uniform: all lanes read same word
      float x1 = b[i * 4], y1 = b[i * 4 + 1], x2 = b[i * 4 + 2], y2 = b[i * 4 + 3];
      float ai = fmaxf(x2 - x1, 0.f) * fmaxf(y2 - y1, 0.f);
      for (int j = i + 1 + tid; j < PRE_NMS_N; j += 1024) {
        if (!keep[j]) continue;
        float u1 = b[j * 4], v1 = b[j * 4 + 1], u2 = b[j * 4 + 2], v2 = b[j * 4 + 3];
        float aj = fmaxf(u2 - u1, 0.f) * fmaxf(v2 - v1, 0.f);
        float iw = fmaxf(fminf(x2, u2) - fmaxf(x1, u1), 0.f);
        float ih = fmaxf(fminf(y2, v2) - fmaxf(y1, v1), 0.f);
        float inter = iw * ih;
        float iou = inter / (ai + aj - inter + 1e-12f);
        if (iou > 0.7f) keep[j] = 0;
      }
    }
    __syncthreads();
  }
  if (tid == 0) {                        // compact: kept first (sorted order), then rest
    int cnt = 0;
    for (int i = 0; i < PRE_NMS_N && cnt < POST_NMS_N; ++i)
      if (keep[i]) {
        for (int c = 0; c < 4; ++c) rois[cnt * 4 + c] = b[i * 4 + c] * 37.f;
        ++cnt;
      }
    for (int i = 0; i < PRE_NMS_N && cnt < POST_NMS_N; ++i)
      if (!keep[i]) {
        for (int c = 0; c < 4; ++c) rois[cnt * 4 + c] = b[i * 4 + c] * 37.f;
        ++cnt;
      }
  }
}

// -------------------------------------------------- RoI max-pool 7x7 -> padded A [304,25088]
__global__ void roipool_kernel(const float* __restrict__ F,
                               const float* __restrict__ rois,
                               float* __restrict__ A) {
  int r = blockIdx.x;
  if (r >= POST_NMS_N) {                 // zero pad rows 300..303 for WMMA M tiling
    for (int idx = threadIdx.x; idx < KFC1; idx += 256) A[(size_t)r * KFC1 + idx] = 0.f;
    return;
  }
  int xs = (int)rintf(rois[r * 4 + 0]);
  int ys = (int)rintf(rois[r * 4 + 1]);
  int xe = (int)rintf(rois[r * 4 + 2]);
  int ye = (int)rintf(rois[r * 4 + 3]);
  int rw = max(xe - xs + 1, 1), rh = max(ye - ys + 1, 1);
  float bw = (float)rw / 7.f, bh = (float)rh / 7.f;
  for (int idx = threadIdx.x; idx < KFC1; idx += 256) {
    int c = idx / 49;
    int rem = idx % 49;
    int ph = rem / 7, pw = rem % 7;
    int hs = min(max((int)floorf(ph * bh) + ys, 0), FH);
    int he = min(max((int)ceilf((ph + 1) * bh) + ys, 0), FH);
    int ws = min(max((int)floorf(pw * bw) + xs, 0), FW);
    int we = min(max((int)ceilf((pw + 1) * bw) + xs, 0), FW);
    float m = -INFINITY;
    for (int h = hs; h < he; ++h)
      for (int w = ws; w < we; ++w)
        m = fmaxf(m, F[(size_t)c * NPIX + h * FW + w]);
    A[(size_t)r * KFC1 + idx] = (m > -1e30f) ? m : 0.f;   // empty bin -> 0
  }
}

// ------------------------------------------------------------- bf16 WMMA GEMM: C = A*W^T + b
// Native fptrunc pair -> should lower to v_cvt_pk_bf16_f32 (hardware RNE)
__device__ __forceinline__ unsigned pack2bf(float f0, float f1) {
  union { __bf16 h[2]; unsigned u; } p;
  p.h[0] = (__bf16)f0;
  p.h[1] = (__bf16)f1;
  return p.u;
}

union FragBF { unsigned u[8]; v16bf v; };

// block = 256 thr (8 waves). WG tile: 16 rows x 256 cols, K-step 64.
// Each wave owns two 16x16 WMMA tiles -> 4 v_wmma per barrier pair.
__global__ __launch_bounds__(256) void gemm_bf16_kernel(const float* __restrict__ A,
                                                        const float* __restrict__ W,
                                                        const float* __restrict__ bias,
                                                        float* __restrict__ C,
                                                        int N, int K, int relu) {
  __shared__ unsigned sA[16 * 32];      // 16 x 64 bf16, [m][k] pairs
  __shared__ unsigned sB[256 * 32];     // 256 x 64 bf16, [n][k] pairs (n-major)
  int tid = threadIdx.x;
  int lane = tid & 31;
  int wv = tid >> 5;
  int m0 = blockIdx.x * 16;
  int n0 = blockIdx.y * 256;
  v8f acc0, acc1;
#pragma unroll
  for (int i = 0; i < 8; ++i) { acc0[i] = 0.f; acc1[i] = 0.f; }
  int mrow = lane & 15, kh = lane >> 4;
  int nc0 = wv * 32 + (lane & 15);      // wave's first N column (local)
  int nc1 = nc0 + 16;
  uint2* sA64 = (uint2*)sA;
  uint2* sB64 = (uint2*)sB;

  for (int k0 = 0; k0 < K; k0 += 64) {
    {                                    // stage A tile 16x64: one float4 per thread
      int m = tid >> 4, kq = tid & 15;   // kq = float4 index within row
      const float4 v4 = *(const float4*)&A[(size_t)(m0 + m) * K + k0 + kq * 4];
      uint2 pk = { pack2bf(v4.x, v4.y), pack2bf(v4.z, v4.w) };
      sA64[m * 16 + kq] = pk;            // u32 idx m*32 + kq*2
    }
#pragma unroll
    for (int rep = 0; rep < 16; ++rep) { // stage B tile 256x64: 16 float4 per thread
      int f = tid + rep * 256;
      int n = f >> 4, kq = f & 15;
      const float4 v4 = *(const float4*)&W[(size_t)(n0 + n) * K + k0 + kq * 4];
      uint2 pk = { pack2bf(v4.x, v4.y), pack2bf(v4.z, v4.w) };
      sB64[n * 16 + kq] = pk;
    }
    if (k0 + 64 < K)                     // global_prefetch_b8 next weight K-slice
      __builtin_prefetch(&W[(size_t)(n0 + tid) * K + k0 + 64], 0, 1);
    __syncthreads();

#pragma unroll
    for (int kc = 0; kc < 2; ++kc) {     // two 32-wide K chunks
      int kb = kc * 16;                  // u32 offset of chunk within row (32 u32/row)
      FragBF fa, fb0, fb1;
#pragma unroll
      for (int v = 0; v < 8; ++v)
        fa.u[v] = sA[mrow * 32 + kb + kh * 8 + v];
#pragma unroll
      for (int v = 0; v < 8; ++v)
        fb0.u[v] = sB[nc0 * 32 + kb + kh * 8 + v];
#pragma unroll
      for (int v = 0; v < 8; ++v)
        fb1.u[v] = sB[nc1 * 32 + kb + kh * 8 + v];
      acc0 = __builtin_amdgcn_wmma_f32_16x16x32_bf16(false, fa.v, false, fb0.v,
                                                     (short)0, acc0, false, false);
      acc1 = __builtin_amdgcn_wmma_f32_16x16x32_bf16(false, fa.v, false, fb1.v,
                                                     (short)0, acc1, false, false);
    }
    __syncthreads();
  }

  int ng0 = n0 + nc0;
  int ng1 = n0 + nc1;
  float bn0 = bias[ng0];
  float bn1 = bias[ng1];
#pragma unroll
  for (int i = 0; i < 8; ++i) {          // C/D layout: VGPR i -> row i (+8 for hi lanes)
    int mg = m0 + i + 8 * kh;
    float v0 = acc0[i] + bn0;
    float v1 = acc1[i] + bn1;
    if (relu) { v0 = fmaxf(v0, 0.f); v1 = fmaxf(v1, 0.f); }
    C[(size_t)mg * N + ng0] = v0;
    C[(size_t)mg * N + ng1] = v1;
  }
}

// -------------------------------------------------------------- output heads (84 reg + 21 cls)
__global__ void heads_kernel(const float* __restrict__ H2,
                             const float* __restrict__ Wr, const float* __restrict__ Br,
                             const float* __restrict__ Wc, const float* __restrict__ Bc,
                             float* __restrict__ out) {
  __shared__ float row[NFC];
  int r = blockIdx.x;
  for (int i = threadIdx.x; i < NFC; i += 128) row[i] = H2[(size_t)r * NFC + i];
  __syncthreads();
  int o = threadIdx.x;
  if (o < 105) {
    const float* w; float b;
    if (o < 84) { w = Wr + (size_t)o * NFC; b = Br[o]; }
    else        { w = Wc + (size_t)(o - 84) * NFC; b = Bc[o - 84]; }
    float acc = b;
    for (int i = 0; i < NFC; ++i) acc = fmaf(row[i], w[i], acc);
    out[r * 105 + o] = acc;
  }
}

// -------------------------------------------------------------------------------- launcher
extern "C" void kernel_launch(void* const* d_in, const int* in_sizes, int n_in,
                              void* d_out, int out_size, void* d_ws, size_t ws_size,
                              hipStream_t stream) {
  const float* F       = (const float*)d_in[0];
  const float* w_inter = (const float*)d_in[1];
  const float* b_inter = (const float*)d_in[2];
  const float* w_cls   = (const float*)d_in[3];
  const float* b_cls   = (const float*)d_in[4];
  const float* w_reg   = (const float*)d_in[5];
  const float* b_reg   = (const float*)d_in[6];
  const float* w_fc1   = (const float*)d_in[7];
  const float* b_fc1   = (const float*)d_in[8];
  const float* w_fc2   = (const float*)d_in[9];
  const float* b_fc2   = (const float*)d_in[10];
  const float* w_clsh  = (const float*)d_in[11];
  const float* b_clsh  = (const float*)d_in[12];
  const float* w_regh  = (const float*)d_in[13];
  const float* b_regh  = (const float*)d_in[14];

  char* ws = (char*)d_ws;
  size_t off = 0;
  auto alloc = [&](size_t bytes) {
    char* p = ws + off;
    off = (off + bytes + 255) & ~(size_t)255;
    return p;
  };
  float* X      = (float*)alloc((size_t)CIN * NPIX * 4);       // conv output
  float* boxes  = (float*)alloc((size_t)NANCH * 4 * 4);
  float* scores = (float*)alloc((size_t)NANCH * 4);
  float* skey   = (float*)alloc((size_t)NSORT * 4);
  int*   sval   = (int*)  alloc((size_t)NSORT * 4);
  float* sbox   = (float*)alloc((size_t)PRE_NMS_N * 4 * 4);
  int*   keep   = (int*)  alloc((size_t)PRE_NMS_N * 4);
  float* rois   = (float*)alloc((size_t)POST_NMS_N * 4 * 4);
  float* Apool  = (float*)alloc((size_t)MPAD * KFC1 * 4);      // pooled features (padded M)
  float* H1     = (float*)alloc((size_t)MPAD * NFC * 4);
  float* H2     = (float*)alloc((size_t)MPAD * NFC * 4);

  conv3x3_relu_kernel<<<NPIX, 256, 0, stream>>>(F, w_inter, b_inter, X);
  rpn_kernel<<<NPIX, 64, 0, stream>>>(X, w_cls, b_cls, w_reg, b_reg, boxes, scores);
  sort_kernel<<<1, 1024, 0, stream>>>(scores, boxes, skey, sval, sbox);
  nms_kernel<<<1, 1024, 0, stream>>>(sbox, keep, rois);
  roipool_kernel<<<MPAD, 256, 0, stream>>>(F, rois, Apool);
  dim3 gg(MPAD / 16, NFC / 256);
  gemm_bf16_kernel<<<gg, 256, 0, stream>>>(Apool, w_fc1, b_fc1, H1, NFC, KFC1, 1);
  gemm_bf16_kernel<<<gg, 256, 0, stream>>>(H1, w_fc2, b_fc2, H2, NFC, NFC, 1);
  heads_kernel<<<POST_NMS_N, 128, 0, stream>>>(H2, w_regh, b_regh, w_clsh, b_clsh, (float*)d_out);
}
